// KnowledgeConsistentAttention_5720896438664
// MI455X (gfx1250) — matrix operand order repaired and checked by reading, not compiled, not run on hardware
//
#include <hip/hip_runtime.h>

typedef unsigned short u16;
typedef __attribute__((ext_vector_type(16))) __bf16 v16bf;
typedef __attribute__((ext_vector_type(8)))  float  v8f;

// ---------------------------------------------------------------------------
// helpers
// ---------------------------------------------------------------------------
__device__ __forceinline__ u16 f32_to_bf16(float f) {
    union { float f; unsigned int u; } v; v.f = f;
    unsigned int r = v.u + 0x7FFFu + ((v.u >> 16) & 1u);   // round-to-nearest-even
    return (u16)(r >> 16);
}

struct U32x8 { uint4 lo, hi; };

__device__ __forceinline__ v16bf ld_frag(const u16* b0, const u16* b1) {
    U32x8 t;
    t.lo = *(const uint4*)b0;
    t.hi = *(const uint4*)b1;
    return __builtin_bit_cast(v16bf, t);
}

// generic pointer -> 32-bit LDS byte offset (addrspace(3) value)
typedef __attribute__((address_space(3))) unsigned char lds_byte_t;
__device__ __forceinline__ unsigned lds_off(const void* p) {
    return (unsigned)(unsigned long long)(lds_byte_t*)p;
}

// CDNA5 direct global -> LDS async copy, 16B per lane, tracked by ASYNCcnt
__device__ __forceinline__ void async_b128(unsigned lds, const u16* g) {
    asm volatile("global_load_async_to_lds_b128 %0, %1, off"
                 :: "v"(lds), "v"((unsigned long long)g) : "memory");
}

// ---------------------------------------------------------------------------
// prep 1: rs -> replicate-pad (65x65) -> conv zero-pad (69x69), NHWC bf16
// rs_pad[n][y][x][c]; y,x in 0..68. rs_p[i] = rs[max(i-1,0)].
// ---------------------------------------------------------------------------
__global__ void prep_rs_pad(const float* __restrict__ rs, u16* __restrict__ rs_pad) {
    const int n = blockIdx.y;
    const int y = blockIdx.x / 69, x = blockIdx.x % 69;
    const int c = threadIdx.x;                     // 256 channels
    float v = 0.0f;
    const int iy = y - 2, ix = x - 2;
    if (iy >= 0 && iy < 65 && ix >= 0 && ix < 65) {
        const int ry = max(iy - 1, 0), rx = max(ix - 1, 0);
        v = rs[((((size_t)n * 256 + c) * 64) + ry) * 64 + rx];
    }
    rs_pad[((((size_t)n * 69 + y) * 69) + x) * 256 + c] = f32_to_bf16(v);
}

// ---------------------------------------------------------------------------
// prep 2: normalized correlation filters  krs[p][tap][c], bf16, p in 0..675
// ---------------------------------------------------------------------------
__global__ void prep_krs(const float* __restrict__ rs, u16* __restrict__ krs) {
    const int p  = blockIdx.x;                     // 676 patches
    const int n  = p / 169, pi = p % 169;
    const int py = pi / 13, px = pi % 13;
    const int c  = threadIdx.x;                    // 256 channels
    const float* base = rs + ((size_t)n * 256 + c) * 4096;

    float v[25];
    float ss = 0.0f;
#pragma unroll
    for (int t = 0; t < 25; ++t) {
        const int dy = t / 5, dx = t % 5;
        const int y = py * 5 + dy, x = px * 5 + dx;          // rs_p coords
        const int ry = max(y - 1, 0), rx = max(x - 1, 0);
        const float f = base[ry * 64 + rx] + 1e-7f;
        v[t] = f;
        ss += f * f;
    }
    __shared__ float red[256];
    red[c] = ss;
    __syncthreads();
    for (int s = 128; s > 0; s >>= 1) {
        if (c < s) red[c] += red[c + s];
        __syncthreads();
    }
    const float inv = rsqrtf(red[0]);
#pragma unroll
    for (int t = 0; t < 25; ++t)
        krs[(((size_t)p * 25) + t) * 256 + c] = f32_to_bf16(v[t] * inv);
}

// ---------------------------------------------------------------------------
// prep 3: paste weights w2[c][tap][p], flipped kernel, p padded 676->704 (zeros)
// w2[c][t=dy*5+dx][p] = k_dem[p][c][4-dy][4-dx] = dem_p[...] + 1e-7
// ---------------------------------------------------------------------------
__global__ void prep_w2(const float* __restrict__ dem, u16* __restrict__ w2) {
    const int t  = blockIdx.x;                     // 25 taps
    const int cc = blockIdx.y;                     // 256 output channels
    const int dy = t / 5, dx = t % 5;
    const int fy = 4 - dy, fx = 4 - dx;
    for (int p = threadIdx.x; p < 704; p += 256) {
        float v = 0.0f;
        if (p < 676) {
            const int n = p / 169, pi = p % 169;
            const int py = pi / 13, px = pi % 13;
            const int y = py * 5 + fy, x = px * 5 + fx;
            const int ry = max(y - 1, 0), rx = max(x - 1, 0);
            v = dem[((((size_t)n * 256 + cc) * 64) + ry) * 64 + rx] + 1e-7f;
        }
        w2[(((size_t)cc * 25) + t) * 704 + p] = f32_to_bf16(v);
    }
}

// ---------------------------------------------------------------------------
// implicit-GEMM 5x5 conv via bf16 WMMA, f32 accumulate.
//   A[m][k] = Apad[n][oy+dy][ox+dx][c]   (NHWC, KC channels, conv-padded)
//   B[p][k] = Bmat[p][tap][c]            (KC contiguous per (p,tap))
//   C[m][p] = Cout[n][m][p]              (pitch NV)
// Block tile 128x128, 8 waves (4 in M x 2 in N), each wave 2x4 fragments
// => 8 x v_wmma_f32_16x16x32_bf16 per 32-wide K step.
// Staging: double-buffered LDS filled with global_load_async_to_lds_b128
// (ASYNCcnt pipeline, 4 copies/thread/step, branch-free via row clamping:
//  clamped rows only feed C rows/cols that the guarded store discards).
// ---------------------------------------------------------------------------
template <int KC, int NV>
__global__ __launch_bounds__(256) void conv_gemm(const u16* __restrict__ Apad,
                                                 const u16* __restrict__ Bmat,
                                                 float* __restrict__ Cout) {
    constexpr int KB    = KC / 32;                 // channel blocks per tap
    constexpr int NSTEP = 25 * KB;                 // total K steps
    const int n    = blockIdx.z;
    const int m0   = blockIdx.x * 128;
    const int p0   = blockIdx.y * 128;
    const int tid  = threadIdx.x;
    const int lane = tid & 31;
    const int wid  = tid >> 5;
    const int wm   = wid & 3;                      // wave M sub-tile (32 rows)
    const int wn   = wid >> 2;                     // wave N sub-tile (64 cols)
    const int g    = lane >> 4;                    // half-wave group
    const int lrow = lane & 15;

    __shared__ u16 sA[2][128 * 32];                // 2 x 8 KB
    __shared__ u16 sB[2][128 * 32];                // 2 x 8 KB

    // ---- per-thread staging geometry: rows (r, r+64), 16B chunk q
    const int r = tid >> 2;                        // 0..63
    const int q = tid & 3;                         // 0..3
    const u16* arow[2];
#pragma unroll
    for (int i = 0; i < 2; ++i) {
        int m = m0 + r + 64 * i;
        if (m > 4224) m = 4224;                    // clamp: feeds discarded rows only
        const int oy = m / 65, ox = m - oy * 65;
        arow[i] = Apad + (((size_t)n * 69 + oy) * 69 + ox) * KC + q * 8;
    }
    const u16* brow[2];
#pragma unroll
    for (int i = 0; i < 2; ++i) {
        int p = p0 + r + 64 * i;
        if (p > NV - 1) p = NV - 1;                // clamp: feeds discarded cols only
        brow[i] = Bmat + (size_t)p * 25 * KC + q * 8;
    }
    const unsigned ldsA = lds_off(&sA[0][0]) + (unsigned)(r * 32 + q * 8) * 2;
    const unsigned ldsB = lds_off(&sB[0][0]) + (unsigned)(r * 32 + q * 8) * 2;

    auto issue = [&](int s, int buf) {
        const int t  = s / KB;
        const int cb = s - t * KB;
        const int dy = t / 5, dx = t - dy * 5;
        const size_t aoff = (size_t)(dy * 69 + dx) * KC + cb * 32;
        const size_t boff = (size_t)t * KC + cb * 32;
        const unsigned bo = (unsigned)buf * (128 * 32 * 2);
        async_b128(ldsA + bo,            arow[0] + aoff);
        async_b128(ldsA + bo + 64 * 64,  arow[1] + aoff);   // row r+64 (64B/row)
        async_b128(ldsB + bo,            brow[0] + boff);
        async_b128(ldsB + bo + 64 * 64,  brow[1] + boff);
    };

    v8f acc[2][4] = {};

    issue(0, 0);
    for (int s = 0; s < NSTEP; ++s) {
        const int cur = s & 1;
        if (s + 1 < NSTEP) {
            issue(s + 1, cur ^ 1);
            // async loads complete in order: <=4 outstanding => step s landed
            asm volatile("s_wait_asynccnt 0x4" ::: "memory");
        } else {
            asm volatile("s_wait_asynccnt 0x0" ::: "memory");
        }
        __syncthreads();                           // all waves' copies visible

        // ---- fragments per ISA 16-bit layouts
        v16bf afrag[2], bfrag[4];
#pragma unroll
        for (int i = 0; i < 2; ++i) {
            const int row = wm * 32 + i * 16 + lrow;          // M row
            // A 16x32: lanes 0-15 K=[0..7,16..23], lanes 16-31 K=[8..15,24..31]
            afrag[i] = ld_frag(&sA[cur][row * 32 + g * 8],
                               &sA[cur][row * 32 + 16 + g * 8]);
        }
#pragma unroll
        for (int j = 0; j < 4; ++j) {
            const int row = wn * 64 + j * 16 + lrow;          // N column
            // B 32x16: lane%16 = N, lanes 0-15 K=[0..7,16..23], lanes 16-31 K=[8..15,24..31]
            bfrag[j] = ld_frag(&sB[cur][row * 32 + g * 8],
                               &sB[cur][row * 32 + 16 + g * 8]);
        }
#pragma unroll
        for (int i = 0; i < 2; ++i)
#pragma unroll
            for (int j = 0; j < 4; ++j)
                acc[i][j] = __builtin_amdgcn_wmma_f32_16x16x32_bf16(
                    false, afrag[i], false, bfrag[j],
                    (short)0, acc[i][j], false, false);

        __syncthreads();                           // safe to overwrite buf cur
    }

    // ---- store C: lane l, vgpr r -> M = r + 8*(l/16), N = l%16
#pragma unroll
    for (int i = 0; i < 2; ++i) {
#pragma unroll
        for (int j = 0; j < 4; ++j) {
            const int col = p0 + wn * 64 + j * 16 + lrow;
            if (col < NV) {
                const int mb = m0 + wm * 32 + i * 16 + 8 * g;
#pragma unroll
                for (int rr = 0; rr < 8; ++rr) {
                    const int m = mb + rr;
                    if (m < 4225)
                        Cout[((size_t)n * 4225 + m) * NV + col] = acc[i][j][rr];
                }
            }
        }
    }
}

// ---------------------------------------------------------------------------
// softmax over 676 filters per (n, position); writes conv-padded NHWC bf16
// att[n][yy][xx][p], p padded to 704 with zeros, border (pad=2) zeroed.
// ---------------------------------------------------------------------------
__global__ void softmax_pad(const float* __restrict__ scores, u16* __restrict__ att) {
    const int n  = blockIdx.y;
    const int yy = blockIdx.x / 69, xx = blockIdx.x % 69;
    const int tid = threadIdx.x;
    u16* dst = att + ((((size_t)n * 69 + yy) * 69) + xx) * 704;

    const int iy = yy - 2, ix = xx - 2;
    if (iy < 0 || iy >= 65 || ix < 0 || ix >= 65) {       // block-uniform branch
        for (int p = tid; p < 704; p += 256) dst[p] = 0;
        return;
    }
    const float* row = scores + ((size_t)n * 4225 + iy * 65 + ix) * 676;

    __shared__ float red[256];
    __shared__ float ev[676];

    float mx = -3.4e38f;
    for (int p = tid; p < 676; p += 256) mx = fmaxf(mx, row[p]);
    red[tid] = mx;
    __syncthreads();
    for (int s = 128; s > 0; s >>= 1) {
        if (tid < s) red[tid] = fmaxf(red[tid], red[tid + s]);
        __syncthreads();
    }
    mx = red[0];
    __syncthreads();

    float sum = 0.0f;
    for (int p = tid; p < 676; p += 256) {
        const float e = __expf(row[p] - mx);
        ev[p] = e;
        sum += e;
    }
    red[tid] = sum;
    __syncthreads();
    for (int s = 128; s > 0; s >>= 1) {
        if (tid < s) red[tid] += red[tid + s];
        __syncthreads();
    }
    const float inv = 1.0f / red[0];
    for (int p = tid; p < 676; p += 256) dst[p] = f32_to_bf16(ev[p] * inv);
    for (int p = 676 + tid; p < 704; p += 256) dst[p] = 0;
}

// ---------------------------------------------------------------------------
// crop [1:65,1:65] + mask blend, NCHW f32 output
// ---------------------------------------------------------------------------
__global__ void blend(const float* __restrict__ fmb, const float* __restrict__ dem,
                      const float* __restrict__ masks, float* __restrict__ out) {
    const size_t idx = (size_t)blockIdx.x * 256 + threadIdx.x;   // 4*256*64*64
    const int x = idx & 63;
    const int y = (idx >> 6) & 63;
    const int c = (idx >> 12) & 255;
    const int n = (int)(idx >> 20);
    const float mk = masks[((size_t)n << 12) + (y << 6) + x];
    const float f  = fmb[((size_t)n * 4225 + (y + 1) * 65 + (x + 1)) * 256 + c];
    out[idx] = f * mk + dem[idx] * (1.0f - mk);
}

// ---------------------------------------------------------------------------
// launch
// ---------------------------------------------------------------------------
extern "C" void kernel_launch(void* const* d_in, const int* in_sizes, int n_in,
                              void* d_out, int out_size, void* d_ws, size_t ws_size,
                              hipStream_t stream) {
    (void)in_sizes; (void)n_in; (void)out_size; (void)ws_size;
    const float* dem   = (const float*)d_in[0];   // (4,256,64,64)
    const float* rs    = (const float*)d_in[1];   // (4,256,64,64)
    const float* masks = (const float*)d_in[2];   // (4,1,64,64)
    float* out = (float*)d_out;                   // (4,256,64,64)

    char* ws = (char*)d_ws;
    u16*   rs_pad = (u16*)(ws);                        //  9,750,528 B
    u16*   krs    = (u16*)(ws + 9750528);              //  8,652,800 B
    u16*   w2     = (u16*)(ws + 18403328);             //  9,011,200 B
    float* scores = (float*)(ws + 27414528);           // 45,697,600 B
    u16*   att    = (u16*)(ws + 73112128);             // 26,813,952 B
    float* fmb    = (float*)(ws + 99926080);           // 17,305,600 B  (end 117,231,680)

    prep_rs_pad<<<dim3(69 * 69, 4), 256, 0, stream>>>(rs, rs_pad);
    prep_krs  <<<676,               256, 0, stream>>>(rs, krs);
    prep_w2   <<<dim3(25, 256),     256, 0, stream>>>(dem, w2);

    // scores[n][m][p]: M=4225, N=676, K=25*256
    conv_gemm<256, 676><<<dim3(34, 6, 4), 256, 0, stream>>>(rs_pad, krs, scores);

    softmax_pad<<<dim3(69 * 69, 4), 256, 0, stream>>>(scores, att);

    // fm[n][m][c]: M=4225, N=256, K=25*704
    conv_gemm<704, 256><<<dim3(34, 2, 4), 256, 0, stream>>>(att, w2, fmb);

    blend<<<(4 * 256 * 64 * 64) / 256, 256, 0, stream>>>(fmb, dem, masks, out);
}
